// LoraConv_79791902425167
// MI455X (gfx1250) — compile-verified, matrix operand order checked by reading, not run on hardware
//
#include <hip/hip_runtime.h>

// ---------------------------------------------------------------------------
// LoRA-MoE conv for MI455X (gfx1250, wave32).
//
// Roofline: conv = 118.5 GFLOP, ~206 MB compulsory HBM traffic (AI ~560
// FLOP/B) -> compute-bound on the vector pipes but near the memory roof on
// the matrix pipes: run it as bf16 WMMA (v_wmma_f32_16x16x32_bf16, f32
// accumulate over K=2304). x (103 MB) is L2-resident (192 MB L2), so halo
// re-reads are cheap; per-expert weights (1.18 MB) are L2-resident too.
//
// Structure:
//  k1: top-1 expert per sample (32 threads).
//  k2: materialize per-expert bf16 weights, tap-major [e][tap][o][ci] so the
//      WMMA A-fragment K-dim (=channels) loads as global b128. (~70 MFLOP)
//  k3: implicit-GEMM conv. Block = (sample b, output row h), M=256 COUT,
//      N=64 cols (56 valid). 8 waves as 4(M)x2(N), wave tile 64x32 = 4x2
//      f32 accumulators. K loop: 8 chunks of 32 channels, double-buffered
//      LDS x-patch (reg-mediated staging overlaps the 72-WMMA burst),
//      9 taps/chunk, each tap a K=32 WMMA step.
// ---------------------------------------------------------------------------

typedef __attribute__((ext_vector_type(16))) __bf16 v16bf;
typedef __attribute__((ext_vector_type(8)))  __bf16 v8bf;
typedef __attribute__((ext_vector_type(8)))  float  v8f;

union ABFrag { v16bf v; v8bf h[2]; };

#define NB    32
#define NE    5
#define CIN   256
#define COUT  256
#define HW    56
#define KCH   32                  // channels staged per K-chunk
#define LCOL  64                  // staged cols (pow2): input col -1..62
#define BUFE  (3 * LCOL * KCH)    // 6144 bf16 per buffer
#define NSTG  24                  // staged elements per thread (6144/256)

// -------------------- kernel 1: top-1 expert per sample ---------------------
__global__ void lora_expert_idx(const float* __restrict__ scores,
                                int* __restrict__ eidx) {
  int b = threadIdx.x;
  if (b < NB) {
    const float* s = scores + b * NE;
    int best = 0; float bv = s[0];
    #pragma unroll
    for (int e = 1; e < NE; ++e) { float v = s[e]; if (v > bv) { bv = v; best = e; } }
    eidx[b] = best;  // first max wins, matches jnp.argmax
  }
}

// ---- kernel 2: Wbf[e][tap][o][ci] = bf16(weight + 4.0 * (B@A) reshaped) ----
// delta[e,o,ci,k1,k2] = sum_r lora_B[e, 3o + t/768, r] * lora_A[e, r, t%768],
// t = ci*9 + tap (tap = k1*3 + k2), SCALING = 16/4 = 4.
__global__ __launch_bounds__(256)
void lora_build_weights(const float* __restrict__ weight,
                        const float* __restrict__ lora_A,
                        const float* __restrict__ lora_B,
                        __bf16* __restrict__ Wbf) {
  int ci  = threadIdx.x;          // 0..255
  int bid = blockIdx.x;           // e*9*256 + tap*256 + o
  int o   = bid % 256;
  int et  = bid / 256;
  int tap = et % 9;
  int e   = et / 9;

  int t    = ci * 9 + tap;
  int rowB = 3 * o + t / 768;
  int colA = t % 768;

  const float* lb = lora_B + ((size_t)e * 768 + rowB) * 12;
  const float* la = lora_A + (size_t)e * 12 * 768 + colA;
  float acc = 0.f;
  #pragma unroll
  for (int r = 0; r < 12; ++r) acc += lb[r] * la[r * 768];

  float wv = weight[((size_t)o * CIN + ci) * 9 + tap] + 4.0f * acc;
  Wbf[(((size_t)e * 9 + tap) * COUT + o) * CIN + ci] = (__bf16)wv;
}

// -------------------- kernel 3: implicit-GEMM conv (WMMA) -------------------
__global__ __launch_bounds__(256)
void lora_conv_wmma(const float* __restrict__ x,
                    const __bf16* __restrict__ Wbf,
                    const int* __restrict__ eidx,
                    float* __restrict__ out) {
  __shared__ __align__(16) __bf16 xs[2 * BUFE];  // double-buffered [r3][c64][ch32]

  const int h  = blockIdx.x;       // 0..55
  const int b  = blockIdx.y;       // 0..31
  const int e  = eidx[b];

  const int tid  = threadIdx.x;
  const int lane = tid & 31;
  const int wid  = tid >> 5;
  const int wm   = wid & 3;        // wave M tile: wm*64
  const int wn   = wid >> 2;       // wave N tile: wn*32
  const int l16  = lane & 15;
  const int hsel = lane >> 4;      // 0: K 0-7/16-23, 1: K 8-15/24-31 (ISA layout)

  // ---- staging geometry (all pow2 -> shifts/ands, col is loop-invariant) ---
  const int jc    = tid & 63;              // staged col 0..63, iw = jc-1
  const int ubase = tid >> 6;              // 0..3
  const int iw    = jc - 1;
  const bool colok = (iw >= 0) & (iw < HW);

  v8f acc[4][2];
  #pragma unroll
  for (int mi = 0; mi < 4; ++mi)
    #pragma unroll
    for (int ni = 0; ni < 2; ++ni)
      acc[mi][ni] = (v8f){0.f, 0.f, 0.f, 0.f, 0.f, 0.f, 0.f, 0.f};

  const __bf16* We = Wbf + (size_t)e * 9 * COUT * CIN;
  const float*  xb = x + (size_t)b * CIN * HW * HW;

  float vreg[NSTG];

  // load one chunk's patch into registers (global f32, coalesced over cols)
  auto load_regs = [&](int ci0) {
    #pragma unroll
    for (int k = 0; k < NSTG; ++k) {
      int u  = ubase + 4 * k;        // 0..95
      int ch = u & (KCH - 1);
      int r  = u >> 5;               // 0..2
      int ih = h - 1 + r;
      float v = 0.f;
      if (colok && ih >= 0 && ih < HW)
        v = xb[(size_t)(ci0 + ch) * (HW * HW) + ih * HW + iw];
      vreg[k] = v;
    }
  };
  // convert + scatter registers into an LDS buffer (channel-fastest)
  auto store_regs = [&](__bf16* buf) {
    #pragma unroll
    for (int k = 0; k < NSTG; ++k) {
      int u  = ubase + 4 * k;
      int ch = u & (KCH - 1);
      int r  = u >> 5;
      buf[(r * LCOL + jc) * KCH + ch] = (__bf16)vreg[k];
    }
  };

  // prologue: stage chunk 0
  load_regs(0);
  store_regs(xs);
  __syncthreads();

  for (int chunk = 0; chunk < CIN / KCH; ++chunk) {
    const int  ci0      = chunk * KCH;
    const bool has_next = (chunk + 1) < (CIN / KCH);

    // issue next chunk's global loads early; they retire under the WMMAs
    if (has_next) load_regs(ci0 + KCH);

    const __bf16* xsb = xs + (chunk & 1) * BUFE;

    #pragma unroll
    for (int tap = 0; tap < 9; ++tap) {
      const int k1 = tap / 3, k2 = tap % 3;

      // A fragments: W'[e][tap][m][ci0 + hsel*8 + {0..7 | 16..23}]  (b128 x2)
      ABFrag afr[4];
      const __bf16* wt = We + (size_t)tap * COUT * CIN;
      #pragma unroll
      for (int mi = 0; mi < 4; ++mi) {
        int o = wm * 64 + mi * 16 + l16;
        const __bf16* pa = wt + (size_t)o * CIN + ci0 + hsel * 8;
        afr[mi].h[0] = *(const v8bf*)(pa);       // K 0-7  / 8-15
        afr[mi].h[1] = *(const v8bf*)(pa + 16);  // K 16-23/ 24-31
      }
      // B fragments from LDS: K = channel, N = output col (lane)
      ABFrag bfr[2];
      #pragma unroll
      for (int ni = 0; ni < 2; ++ni) {
        int w = wn * 32 + ni * 16 + l16;         // 0..63
        int c = w + k2;                          // valid lanes: <= 57
        if (c > LCOL - 1) c = LCOL - 1;          // keep padded lanes in-bounds
        const __bf16* pb = &xsb[(k1 * LCOL + c) * KCH + hsel * 8];
        bfr[ni].h[0] = *(const v8bf*)(pb);       // ds_load_b128
        bfr[ni].h[1] = *(const v8bf*)(pb + 16);
      }
      // 8 WMMAs per tap per wave
      #pragma unroll
      for (int mi = 0; mi < 4; ++mi)
        #pragma unroll
        for (int ni = 0; ni < 2; ++ni)
          acc[mi][ni] = __builtin_amdgcn_wmma_f32_16x16x32_bf16(
              false, afr[mi].v, false, bfr[ni].v,
              (short)0, acc[mi][ni], false, false);
    }

    if (has_next) store_regs(xs + ((chunk + 1) & 1) * BUFE);
    __syncthreads();  // next buffer written; current buffer free
  }

  // Epilogue: C layout — VGPR r: lanes 0-15 M=r, lanes 16-31 M=r+8, N=l16.
  #pragma unroll
  for (int mi = 0; mi < 4; ++mi) {
    #pragma unroll
    for (int ni = 0; ni < 2; ++ni) {
      int w = wn * 32 + ni * 16 + l16;
      if (w < HW) {
        #pragma unroll
        for (int r = 0; r < 8; ++r) {
          int m = wm * 64 + mi * 16 + hsel * 8 + r;
          out[(((size_t)b * COUT + m) * HW + h) * HW + w] = acc[mi][ni][r];
        }
      }
    }
  }
}

// ---------------------------------------------------------------------------
extern "C" void kernel_launch(void* const* d_in, const int* in_sizes, int n_in,
                              void* d_out, int out_size, void* d_ws, size_t ws_size,
                              hipStream_t stream) {
  const float* x      = (const float*)d_in[0];
  const float* scores = (const float*)d_in[1];
  const float* weight = (const float*)d_in[2];
  const float* lora_A = (const float*)d_in[3];
  const float* lora_B = (const float*)d_in[4];
  float* out = (float*)d_out;

  int*    eidx = (int*)d_ws;                         // 32 ints
  __bf16* Wbf  = (__bf16*)((char*)d_ws + 256);       // 5*9*256*256 bf16 = 5.9 MB

  lora_expert_idx<<<1, 32, 0, stream>>>(scores, eidx);
  lora_build_weights<<<NE * 9 * COUT, 256, 0, stream>>>(weight, lora_A, lora_B, Wbf);
  dim3 grid(HW, NB);  // (h, b) -> 1792 workgroups
  lora_conv_wmma<<<grid, 256, 0, stream>>>(x, Wbf, eidx, out);
}
